// GRASSEncoder_41274635714627
// MI455X (gfx1250) — matrix operand order repaired and checked by reading, not compiled, not run on hardware
//
#include <hip/hip_runtime.h>

// ---------------- types ----------------
typedef __attribute__((ext_vector_type(16))) __bf16 v16bf;
typedef __attribute__((ext_vector_type(8)))  __bf16 v8bf;
typedef __attribute__((ext_vector_type(8)))  float  v8f;

__device__ __forceinline__ __bf16 to_bf16(float x) { return (__bf16)x; }

// Fast tanh: prefer CDNA5 hardware v_tanh_f32 (TRANS op); fallback is a
// branch-free exp2/rcp formulation (clamped so exp2 never produces inf/inf).
__device__ __forceinline__ float fast_tanh(float x) {
#if defined(__HIP_DEVICE_COMPILE__)
#if __has_builtin(__builtin_amdgcn_tanhf)
  return __builtin_amdgcn_tanhf(x);
#else
  float xc = fminf(fmaxf(x, -10.f), 10.f);
  float t  = __builtin_amdgcn_exp2f(xc * 2.8853900817779268f);  // exp(2x)
  return (t - 1.f) * __builtin_amdgcn_rcpf(t + 1.f);
#endif
#else
  return tanhf(x);
#endif
}

// ---------------- problem constants ----------------
#define BATCH 2048
#define NBOX  64
#define NSYM  16
#define BOXD  12
#define SYMD  8
#define FD    512
#define HD    1024

// ---------------- GEMM tiling ----------------
#define TM 64
#define TN 64
#define TK 32
#define LDSS 40      // padded row stride (bf16 elems) to avoid LDS bank conflicts

// ============================================================
// Weight conversion / padding (run once per call, fully rewrites
// its outputs so the 0xAA-poisoned workspace never leaks through).
// ============================================================
__global__ void grass_convert_kernel(
    const float* __restrict__ Wl, const float* __restrict__ Wr,
    const float* __restrict__ Ws, const float* __restrict__ Sl,
    const float* __restrict__ Ss, const float* __restrict__ Sr,
    const float* __restrict__ sym, const float* __restrict__ sbl,
    const float* __restrict__ sbr,
    __bf16* __restrict__ WlB, __bf16* __restrict__ WrB, __bf16* __restrict__ WsB,
    __bf16* __restrict__ SlB, __bf16* __restrict__ SsB,
    __bf16* __restrict__ SrPad, __bf16* __restrict__ symPad,
    float* __restrict__ sblr)
{
  long i = (long)blockIdx.x * 256 + threadIdx.x;
  const long W = (long)HD * FD;                        // 524288
  if (i < W) { WlB[i] = to_bf16(Wl[i]); return; } i -= W;
  if (i < W) { WrB[i] = to_bf16(Wr[i]); return; } i -= W;
  if (i < W) { WsB[i] = to_bf16(Ws[i]); return; } i -= W;
  if (i < W) { SlB[i] = to_bf16(Sl[i]); return; } i -= W;
  if (i < W) { SsB[i] = to_bf16(Ss[i]); return; } i -= W;
  if (i < (long)HD * 32) {                             // Sr: [1024][8] -> [1024][32] zero-pad
    long n = i >> 5; int k = (int)(i & 31);
    SrPad[i] = to_bf16(k < SYMD ? Sr[n * SYMD + k] : 0.f);
    return;
  }
  i -= (long)HD * 32;
  if (i < (long)BATCH * NSYM * 32) {                   // sym: [2048][16][8] -> [2048][16][32]
    long q = i >> 5; int k = (int)(i & 31);
    symPad[i] = to_bf16(k < SYMD ? sym[q * SYMD + k] : 0.f);
    return;
  }
  i -= (long)BATCH * NSYM * 32;
  if (i < HD) sblr[i] = sbl[i] + sbr[i];
}

// ============================================================
// Box projection: out[m][n] = tanh(input[m][kbox][:] @ Wb[n][:] + bb[n])
// K=12 -> plain fp32 VALU (0.3% of total FLOPs), bf16 output.
// ============================================================
__global__ __launch_bounds__(256)
void grass_box_kernel(const float* __restrict__ in, const float* __restrict__ Wb,
                      const float* __restrict__ bb, __bf16* __restrict__ out, int kbox)
{
  long idx = (long)blockIdx.x * 256 + threadIdx.x;   // BATCH*FD threads
  int n = (int)(idx & (FD - 1));
  long m = idx >> 9;                                  // /512
  const float* row = in + (m * NBOX + kbox) * BOXD;
  const float* w = Wb + (long)n * BOXD;
  float acc = bb[n];
#pragma unroll
  for (int q = 0; q < BOXD; ++q) acc += row[q] * w[q];
  out[m * FD + n] = to_bf16(fast_tanh(acc));
}

// ============================================================
// Dual-A bf16 WMMA GEMM:
//   out[m][n] = tanh( sum_k A1[m][k]*W1[n][k] (k<K1)
//                   + sum_k A2[m][k]*W2[n][k] (k<K2)  + bias[n] )
// M = 2048 (grid.y*64), N = grid.x*64. K1,K2 multiples of 32 (K2 may be 0).
// 128 threads = 4 wave32; each wave owns a 32x32 quadrant = 2x2 WMMA tiles.
// Bias is splatted into the accumulator init (free epilogue bias add).
// ============================================================
__global__ __launch_bounds__(128)
void grass_wmma_gemm(const __bf16* __restrict__ A1, int lda1,
                     const __bf16* __restrict__ W1, int K1,
                     const __bf16* __restrict__ A2, int lda2,
                     const __bf16* __restrict__ W2, int K2, int ldw2,
                     const float* __restrict__ bias,
                     __bf16* __restrict__ outB, float* __restrict__ outF, int ldo)
{
  __shared__ __align__(16) __bf16 ldsA[TM * LDSS];
  __shared__ __align__(16) __bf16 ldsB[TN * LDSS];

  const int tid  = threadIdx.x;
  const int lane = tid & 31;
  const int wave = tid >> 5;
  const int m0 = blockIdx.y * TM;
  const int n0 = blockIdx.x * TN;
  const int wm = (wave >> 1) * 32;     // wave quadrant inside tile
  const int wn = (wave & 1) * 32;
  const int lrow = lane & 15;
  const int hi   = lane >> 4;          // lane half selects K sub-blocks

  // Accumulator init = bias broadcast: C lane layout is N = lane&15 for every
  // VGPR, so splatting bias[n] across the v8f is exactly the right bias add.
  const float bj0 = bias[n0 + wn + 0 * 16 + lrow];
  const float bj1 = bias[n0 + wn + 1 * 16 + lrow];
  v8f c[2][2];
  c[0][0] = bj0; c[1][0] = bj0;
  c[0][1] = bj1; c[1][1] = bj1;

  const int totalChunks = (K1 + K2) / TK;
  for (int ch = 0; ch < totalChunks; ++ch) {
    const int k = ch * TK;
    const __bf16* Ap; const __bf16* Wp; int lda, ldw, k0;
    if (k < K1) { Ap = A1; Wp = W1; lda = lda1; ldw = K1;  k0 = k; }
    else        { Ap = A2; Wp = W2; lda = lda2; ldw = ldw2; k0 = k - K1; }

    // Stage 64x32 A-tile and 64x32 B-tile (W is [N][K] row-major already).
    // 256 chunks of 8 bf16 (16B) each, 2 per thread per tile.
#pragma unroll
    for (int it = 0; it < 2; ++it) {
      int i   = tid + it * 128;
      int row = i >> 2;
      int col = (i & 3) * 8;
      *(v8bf*)&ldsA[row * LDSS + col] = *(const v8bf*)&Ap[(long)(m0 + row) * lda + k0 + col];
      *(v8bf*)&ldsB[row * LDSS + col] = *(const v8bf*)&Wp[(long)(n0 + row) * ldw + k0 + col];
    }
    __syncthreads();

    // A fragments: lane half 0 -> K 0..7 & 16..23, half 1 -> K 8..15 & 24..31
    v16bf a[2], b[2];
#pragma unroll
    for (int i = 0; i < 2; ++i) {
      int m = wm + i * 16 + lrow;
      union { v16bf v; v8bf h[2]; } u;
      u.h[0] = *(const v8bf*)&ldsA[m * LDSS + hi * 8];
      u.h[1] = *(const v8bf*)&ldsA[m * LDSS + hi * 8 + 16];
      a[i] = u.v;
    }
    // B fragments: lane half 0 -> K 0..7 & 16..23, half 1 -> K 8..15 & 24..31
#pragma unroll
    for (int j = 0; j < 2; ++j) {
      int n = wn + j * 16 + lrow;
      union { v16bf v; v8bf h[2]; } u;
      u.h[0] = *(const v8bf*)&ldsB[n * LDSS + hi * 8];
      u.h[1] = *(const v8bf*)&ldsB[n * LDSS + hi * 8 + 16];
      b[j] = u.v;
    }

#pragma unroll
    for (int i = 0; i < 2; ++i)
#pragma unroll
      for (int j = 0; j < 2; ++j)
        c[i][j] = __builtin_amdgcn_wmma_f32_16x16x32_bf16(
            false, a[i], false, b[j], (short)0, c[i][j], false, false);
    __syncthreads();
  }

  // Epilogue: C layout = VGPR r holds row (r + 8*laneHalf), N = lane&15.
  // Branch on output type hoisted out of the store loops; tanh is a single
  // TRANS instruction (v_tanh_f32) when available.
  if (outF) {
#pragma unroll
    for (int i = 0; i < 2; ++i)
#pragma unroll
      for (int j = 0; j < 2; ++j) {
        int n = n0 + wn + j * 16 + lrow;
#pragma unroll
        for (int r = 0; r < 8; ++r) {
          int m = m0 + wm + i * 16 + r + hi * 8;
          outF[(long)m * ldo + n] = fast_tanh(c[i][j][r]);
        }
      }
  } else {
#pragma unroll
    for (int i = 0; i < 2; ++i)
#pragma unroll
      for (int j = 0; j < 2; ++j) {
        int n = n0 + wn + j * 16 + lrow;
#pragma unroll
        for (int r = 0; r < 8; ++r) {
          int m = m0 + wm + i * 16 + r + hi * 8;
          outB[(long)m * ldo + n] = to_bf16(fast_tanh(c[i][j][r]));
        }
      }
  }
}

// ============================================================
// Host-side orchestration (all on `stream`, graph-capture safe)
// ============================================================
extern "C" void kernel_launch(void* const* d_in, const int* in_sizes, int n_in,
                              void* d_out, int out_size, void* d_ws, size_t ws_size,
                              hipStream_t stream)
{
  (void)in_sizes; (void)n_in; (void)out_size; (void)ws_size;

  const float* inputStacks = (const float*)d_in[0];   // [2048][64][12]
  const float* symStacks   = (const float*)d_in[1];   // [2048][16][8]
  // d_in[2] = operations (fixed sequence; ignored)
  const float* Wb  = (const float*)d_in[3];
  const float* bb  = (const float*)d_in[4];
  const float* Wl  = (const float*)d_in[5];
  const float* bl  = (const float*)d_in[6];
  const float* Wr  = (const float*)d_in[7];
  const float* Ws  = (const float*)d_in[8];
  const float* bs  = (const float*)d_in[9];
  const float* Sl  = (const float*)d_in[10];
  const float* sbl = (const float*)d_in[11];
  const float* Sr  = (const float*)d_in[12];
  const float* sbr = (const float*)d_in[13];
  const float* Ss  = (const float*)d_in[14];
  const float* sbs = (const float*)d_in[15];

  // ---- workspace carve-up (256B aligned slots) ----
  char* ws = (char*)d_ws;
  size_t off = 0;
  auto carve = [&](size_t bytes) { size_t o = off; off += (bytes + 255) & ~(size_t)255; return o; };
  __bf16* WlB    = (__bf16*)(ws + carve((size_t)HD * FD * 2));
  __bf16* WrB    = (__bf16*)(ws + carve((size_t)HD * FD * 2));
  __bf16* WsB    = (__bf16*)(ws + carve((size_t)FD * HD * 2));
  __bf16* SlB    = (__bf16*)(ws + carve((size_t)HD * FD * 2));
  __bf16* SsB    = (__bf16*)(ws + carve((size_t)FD * HD * 2));
  __bf16* SrPad  = (__bf16*)(ws + carve((size_t)HD * 32 * 2));
  __bf16* symPad = (__bf16*)(ws + carve((size_t)BATCH * NSYM * 32 * 2));
  float*  sblr   = (float*) (ws + carve((size_t)HD * 4));
  __bf16* boxK   = (__bf16*)(ws + carve((size_t)BATCH * FD * 2));
  __bf16* sBuf   = (__bf16*)(ws + carve((size_t)BATCH * FD * 2));
  __bf16* hBuf   = (__bf16*)(ws + carve((size_t)BATCH * HD * 2));

  // ---- 1) weight conversion / padding ----
  {
    long total = 5L * HD * FD + (long)HD * 32 + (long)BATCH * NSYM * 32 + HD;
    int blocks = (int)((total + 255) / 256);
    grass_convert_kernel<<<blocks, 256, 0, stream>>>(
        Wl, Wr, Ws, Sl, Ss, Sr, symStacks, sbl, sbr,
        WlB, WrB, WsB, SlB, SsB, SrPad, symPad, sblr);
  }

  const int boxBlocks = (BATCH * FD) / 256;
  const dim3 g1(HD / TN, BATCH / TM);   // GEMM -> h  (N=1024)
  const dim3 g2(FD / TN, BATCH / TM);   // GEMM -> s  (N=512)

  // ---- 2) s0 = box[63] ----
  grass_box_kernel<<<boxBlocks, 256, 0, stream>>>(inputStacks, Wb, bb, sBuf, NBOX - 1);

  // ---- 3) 63 reduce steps ----
  for (int j = 0; j < NBOX - 1; ++j) {
    int k = NBOX - 2 - j;  // rights: 62 .. 0
    grass_box_kernel<<<boxBlocks, 256, 0, stream>>>(inputStacks, Wb, bb, boxK, k);
    // h = tanh(s@Wl.T + box_k@Wr.T + bl)
    grass_wmma_gemm<<<g1, 128, 0, stream>>>(
        sBuf, FD, WlB, FD, boxK, FD, WrB, FD, FD, bl, hBuf, nullptr, HD);
    // s = tanh(h@Ws.T + bs)
    grass_wmma_gemm<<<g2, 128, 0, stream>>>(
        hBuf, HD, WsB, HD, nullptr, 0, nullptr, 0, 0, bs, sBuf, nullptr, FD);
  }

  // ---- 4) 16 symmetry steps (t = 15 .. 0) ----
  for (int t = NSYM - 1; t >= 0; --t) {
    // h = tanh(s@Sl.T + sym_t@Sr.T + (sbl+sbr)); sym slice is [2048][32] zero-padded
    grass_wmma_gemm<<<g1, 128, 0, stream>>>(
        sBuf, FD, SlB, FD, symPad + (size_t)t * 32, NSYM * 32, SrPad, 32, 32,
        sblr, hBuf, nullptr, HD);
    // s = tanh(h@Ss.T + sbs); final step writes fp32 straight to d_out
    float* outF = (t == 0) ? (float*)d_out : nullptr;
    grass_wmma_gemm<<<g2, 128, 0, stream>>>(
        hBuf, HD, SsB, HD, nullptr, 0, nullptr, 0, 0, sbs, sBuf, outF, FD);
  }
}
